// ExpertPrefetchHead_72292889526404
// MI455X (gfx1250) — matrix operand order repaired and checked by reading, compiled, not run on hardware
//
#include <hip/hip_runtime.h>

// ---------------- problem constants (from reference) ----------------
#define LL   20      // routing layers
#define HH   2048    // hidden dim
#define RR   512     // proj rank
#define AR   64      // adapter rank
#define EE   256     // experts
#define TOPK 8
#define NTOK 8192    // B*K = 4*2048

typedef __attribute__((ext_vector_type(16))) __bf16 v16bf;
typedef __attribute__((ext_vector_type(8)))  float  v8f;

union AFrag { uint4 q[2]; v16bf v; };

// fp32 -> bf16 round-to-nearest-even
__device__ __forceinline__ unsigned short f2bf(float f) {
    union { float f; unsigned u; } c; c.f = f;
    unsigned u = c.u;
    if ((u & 0x7fffffffu) > 0x7f800000u) return (unsigned short)((u >> 16) | 0x0040u); // quiet NaN
    unsigned r = u + 0x7fffu + ((u >> 16) & 1u);
    return (unsigned short)(r >> 16);
}

__device__ __forceinline__ float gelu_exact(float x) {
    return 0.5f * x * (1.0f + erff(x * 0.70710678118654752f));
}

// ---------------- elementwise convert ----------------
__global__ __launch_bounds__(256) void f32_to_bf16_kernel(const float* __restrict__ in,
                                                          unsigned short* __restrict__ out,
                                                          long n) {
    long i = (long)blockIdx.x * blockDim.x + threadIdx.x;
    if (i < n) out[i] = f2bf(in[i]);
}

// ---------------- WMMA bf16 GEMM:  C[M,N] = op(A[M,K] * B[N,K]^T (+ addend)) ----------------
// Tile: block = 128(M) x 64(N), 8 waves, each wave a 32x32 tile (2x2 WMMA accs), K step 32.
// A fragment layout (16-bit 16x32, ISA 7.12.2): lanes 0-15 row M=l16, K = k0+{0..7,16..23};
//                                               lanes 16-31 row M=l16, K = k0+{8..15,24..31}.
// B fragment layout (16-bit 32x16):             lanes 0-15 col N=l16, K = k0+{0..7,16..23};
//                                               lanes 16-31 col N=l16, K = k0+{8..15,24..31}.
// C/D f32 16x16: VGPR r -> (M = r + 8*half, N = l16).
template<bool GELU, bool ADD, bool OUT_BF16>
__global__ __launch_bounds__(256) void gemm_bf16_nt(const unsigned short* __restrict__ A,
                                                    const unsigned short* __restrict__ B,
                                                    const float* __restrict__ addend,
                                                    void* __restrict__ Cout,
                                                    int M, int N, int K,
                                                    long strideA, long strideB, long strideC) {
    const int zb = blockIdx.z;
    A += (long)zb * strideA;
    B += (long)zb * strideB;

    const int bm   = blockIdx.y * 128;
    const int bn   = blockIdx.x * 64;
    const int wid  = threadIdx.x >> 5;
    const int lane = threadIdx.x & 31;
    const int wm   = (wid & 3) * 32;   // 4 wave-rows
    const int wn   = (wid >> 2) * 32;  // 2 wave-cols
    const int half = lane >> 4;
    const int l16  = lane & 15;

    v8f acc[2][2] = {};

    for (int k0 = 0; k0 < K; k0 += 32) {
        AFrag af[2], bf[2];
#pragma unroll
        for (int i = 0; i < 2; ++i) {
            const long row = bm + wm + i * 16 + l16;
            const unsigned short* ap = A + row * (long)K + k0 + half * 8;
            af[i].q[0] = *(const uint4*)(ap);
            af[i].q[1] = *(const uint4*)(ap + 16);
        }
#pragma unroll
        for (int j = 0; j < 2; ++j) {
            const long col = bn + wn + j * 16 + l16;
            const unsigned short* bp = B + col * (long)K + k0 + half * 8;
            bf[j].q[0] = *(const uint4*)(bp);
            bf[j].q[1] = *(const uint4*)(bp + 16);
        }
#pragma unroll
        for (int i = 0; i < 2; ++i)
#pragma unroll
            for (int j = 0; j < 2; ++j)
                acc[i][j] = __builtin_amdgcn_wmma_f32_16x16x32_bf16(
                    false, af[i].v, false, bf[j].v, (short)0, acc[i][j], false, false);
    }

    // epilogue
#pragma unroll
    for (int i = 0; i < 2; ++i) {
#pragma unroll
        for (int j = 0; j < 2; ++j) {
#pragma unroll
            for (int r = 0; r < 8; ++r) {
                const long row = bm + wm + i * 16 + half * 8 + r;
                const long col = bn + wn + j * 16 + l16;
                float v = acc[i][j][r];
                if (ADD)  v += addend[row * (long)N + col];
                if (GELU) v = gelu_exact(v);
                const long cidx = (long)zb * strideC + row * (long)N + col;
                if (OUT_BF16) ((unsigned short*)Cout)[cidx] = f2bf(v);
                else          ((float*)Cout)[cidx] = v;
            }
        }
    }
}

// ---------------- top-k (k=8) over rows of 256 logits; one wave32 per row ----------------
// Element index layout per lane: idx = j*32 + lane  (j = 0..7). Tie-break: lowest index,
// matching jax.lax.top_k.
__global__ __launch_bounds__(256) void topk_rows_kernel(const float* __restrict__ logits,
                                                        float* __restrict__ out_idx,
                                                        long nrows) {
    const long row = (long)blockIdx.x * 8 + (threadIdx.x >> 5);
    if (row >= nrows) return;
    const int lane = threadIdx.x & 31;
    const float* p = logits + row * (long)EE;

    float v[8];
#pragma unroll
    for (int j = 0; j < 8; ++j) v[j] = p[j * 32 + lane];

    for (int r = 0; r < TOPK; ++r) {
        float bv = v[0];
        int   bi = lane;
#pragma unroll
        for (int j = 1; j < 8; ++j) {
            const int idx = j * 32 + lane;
            if (v[j] > bv) { bv = v[j]; bi = idx; }   // strict > keeps lowest index
        }
#pragma unroll
        for (int off = 16; off > 0; off >>= 1) {
            const float ov = __shfl_xor(bv, off, 32);
            const int   oi = __shfl_xor(bi, off, 32);
            if (ov > bv || (ov == bv && oi < bi)) { bv = ov; bi = oi; }
        }
        if ((bi & 31) == lane) v[bi >> 5] = -__builtin_inff();
        if (lane == 0) out_idx[row * TOPK + r] = (float)bi;
    }
}

// ---------------- host side ----------------
extern "C" void kernel_launch(void* const* d_in, const int* in_sizes, int n_in,
                              void* d_out, int out_size, void* d_ws, size_t ws_size,
                              hipStream_t stream) {
    (void)in_sizes; (void)n_in; (void)out_size; (void)ws_size;

    const float* x   = (const float*)d_in[0];  // [NTOK, H]
    const float* wd  = (const float*)d_in[1];  // [R, H]
    const float* wu  = (const float*)d_in[2];  // [H, R]
    const float* ad  = (const float*)d_in[3];  // [L, A, H]
    const float* au  = (const float*)d_in[4];  // [L, H, A]
    const float* gw  = (const float*)d_in[5];  // [L, E, H]

    float* out = (float*)d_out;
    const long IDX_COUNT = (long)LL * NTOK * TOPK;       // indices first (as floats)
    float* out_logits = out + IDX_COUNT;                 // then [L, NTOK, E] fp32

    // --- workspace carve-up (256B aligned) ---
    char* ws = (char*)d_ws;
    size_t off = 0;
    auto carve = [&](size_t bytes) -> char* {
        char* p = ws + off;
        off = (off + bytes + 255) & ~(size_t)255;
        return p;
    };
    unsigned short* xb    = (unsigned short*)carve((size_t)NTOK * HH * 2);
    unsigned short* wdb   = (unsigned short*)carve((size_t)RR * HH * 2);
    unsigned short* wub   = (unsigned short*)carve((size_t)HH * RR * 2);
    unsigned short* adb   = (unsigned short*)carve((size_t)LL * AR * HH * 2);
    unsigned short* aub   = (unsigned short*)carve((size_t)LL * HH * AR * 2);
    unsigned short* gwb   = (unsigned short*)carve((size_t)LL * EE * HH * 2);
    unsigned short* pbuf  = (unsigned short*)carve((size_t)NTOK * RR * 2);
    float*          shf   = (float*)         carve((size_t)NTOK * HH * 4);
    unsigned short* a_all = (unsigned short*)carve((size_t)LL * NTOK * AR * 2);
    unsigned short* hbuf  = (unsigned short*)carve((size_t)NTOK * HH * 2);

    const dim3 blk(256);
    auto cvt = [&](const float* s, unsigned short* d, long n) {
        f32_to_bf16_kernel<<<dim3((unsigned)((n + 255) / 256)), blk, 0, stream>>>(s, d, n);
    };

    // 1) fp32 -> bf16 staging
    cvt(x,  xb,  (long)NTOK * HH);
    cvt(wd, wdb, (long)RR * HH);
    cvt(wu, wub, (long)HH * RR);
    cvt(ad, adb, (long)LL * AR * HH);
    cvt(au, aub, (long)LL * HH * AR);
    cvt(gw, gwb, (long)LL * EE * HH);

    // 2) P = gelu(x @ Wd^T)            [NTOK, R] bf16
    gemm_bf16_nt<true, false, true><<<dim3(RR / 64, NTOK / 128, 1), blk, 0, stream>>>(
        xb, wdb, nullptr, pbuf, NTOK, RR, HH, 0, 0, 0);

    // 3) shared = P @ Wu^T             [NTOK, H] f32
    gemm_bf16_nt<false, false, false><<<dim3(HH / 64, NTOK / 128, 1), blk, 0, stream>>>(
        pbuf, wub, nullptr, shf, NTOK, HH, RR, 0, 0, 0);

    // 4) a_l = gelu(x @ Ad_l^T), all L layers batched   [L, NTOK, A] bf16
    gemm_bf16_nt<true, false, true><<<dim3(AR / 64, NTOK / 128, LL), blk, 0, stream>>>(
        xb, adb, nullptr, a_all, NTOK, AR, HH,
        0, (long)AR * HH, (long)NTOK * AR);

    // 5) per layer: h = a_l @ Au_l^T + shared (bf16), logits = h @ G_l^T (f32 -> d_out)
    for (int l = 0; l < LL; ++l) {
        gemm_bf16_nt<false, true, true><<<dim3(HH / 64, NTOK / 128, 1), blk, 0, stream>>>(
            a_all + (long)l * NTOK * AR, aub + (long)l * HH * AR, shf, hbuf,
            NTOK, HH, AR, 0, 0, 0);
        gemm_bf16_nt<false, false, false><<<dim3(EE / 64, NTOK / 128, 1), blk, 0, stream>>>(
            hbuf, gwb + (long)l * EE * HH, nullptr, out_logits + (long)l * NTOK * EE,
            NTOK, EE, HH, 0, 0, 0);
    }

    // 6) top-8 indices per (layer, token) row -> first region of d_out (as floats)
    const long nrows = (long)LL * NTOK;
    topk_rows_kernel<<<dim3((unsigned)((nrows + 7) / 8)), blk, 0, stream>>>(
        out_logits, out, nrows);
}
